// Pct_partseg7_7662221656502
// MI455X (gfx1250) — matrix operand order repaired and checked by reading, not compiled, not run on hardware
//
#include <hip/hip_runtime.h>
#include <hip/hip_bf16.h>

typedef __attribute__((ext_vector_type(16))) _Float16 v16h;
typedef __attribute__((ext_vector_type(2)))  __fp16   h2v;   // cvt_pkrtz return type
typedef __attribute__((ext_vector_type(8)))  float    v8f;

#define B8   8
#define NPTS 2048
#define NPOINT 512
#define NSAMPLE 32
#define PART 50

union V16 { v16h v; h2v p[8]; };

// ---------------------------------------------------------------------------
// f16-WMMA GEMM: D[b][m][n] = sum_k A[b][m][k] * B[b][k][n] (+bias[m])
// ACS: compile-time A k-stride (1 = contiguous weights; 2048 = transposed q).
// BRS: compile-time B k-stride (2048 point GEMMs, 32 group conv).
// Compile-time strides -> every fragment load is base + constant immediate
// offset (max 31*2048*4 B < 24-bit IOFFSET), one pointer bump per K-step.
// One wave computes a 16 x (16*NT) tile; K loop: unguarded full 32-tiles +
// clamped tail (only conv1 has K%32 != 0). m/n clamped, not guarded: garbage
// rows/cols only affect D entries we never store.
// Fragment layouts used here (wave32; contiguous k, same order in A and B):
//   A: lane(half h,row r) elem e -> k = 16h + e
//   B: lane(half h,col r) elem e -> k = 16h + e
//   D: lane(half h,col r) reg rr -> m = rr + 8h
// ---------------------------------------------------------------------------
template<int NT, int BRS, int ACS>
__global__ void k_gemm(const float* __restrict__ A, int ars, long long abs_,
                       const float* __restrict__ Bm, long long bbs,
                       float* __restrict__ D, int drs, long long dbs,
                       const float* __restrict__ bias, int M, int N, int K) {
  const int lane = threadIdx.x & 31;
  const int h = lane >> 4;
  const int r = lane & 15;
  const int n0 = blockIdx.x * (16 * NT);
  const int m0 = blockIdx.y * 16;
  const long long b = blockIdx.z;
  const float* Ab = A  + b * abs_;
  const float* Bb = Bm + b * bbs;
  float*       Db = D  + b * dbs;

  const int mC = min(m0 + r, M - 1);                 // clamped row for A loads
  const float* ap0 = Ab + (long long)mC * ars;
  const float* aptr = ap0 + 16 * h * ACS;

  const float* bp0[NT];
  const float* bptr[NT];
#pragma unroll
  for (int t = 0; t < NT; ++t) {
    int nC = min(n0 + 16 * t + r, N - 1);            // clamped col for B loads
    bp0[t]  = Bb + nC;
    bptr[t] = Bb + nC + 16 * h * BRS;
  }

  v8f acc[NT];
#pragma unroll
  for (int t = 0; t < NT; ++t) acc[t] = v8f{0.f,0.f,0.f,0.f,0.f,0.f,0.f,0.f};

  const int kfull = K & ~31;
  for (int k0 = 0; k0 < kfull; k0 += 32) {
    V16 a;
#pragma unroll
    for (int j = 0; j < 4; ++j) {                    // constant immediate offsets
      a.p[j]     = __builtin_amdgcn_cvt_pkrtz(aptr[(2*j)     * ACS], aptr[(2*j + 1)  * ACS]);
      a.p[4 + j] = __builtin_amdgcn_cvt_pkrtz(aptr[(8 + 2*j)* ACS], aptr[(9 + 2*j) * ACS]);
    }
    V16 bf[NT];
#pragma unroll
    for (int t = 0; t < NT; ++t)
#pragma unroll
      for (int j = 0; j < 8; ++j)
        bf[t].p[j] = __builtin_amdgcn_cvt_pkrtz(bptr[t][(2*j) * BRS],
                                                bptr[t][(2*j + 1) * BRS]);
#pragma unroll
    for (int t = 0; t < NT; ++t)
      acc[t] = __builtin_amdgcn_wmma_f32_16x16x32_f16(false, a.v, false, bf[t].v,
                                                      (short)0, acc[t], false, false);
    aptr += 32 * ACS;
#pragma unroll
    for (int t = 0; t < NT; ++t) bptr[t] += 32 * BRS;
  }

  if (kfull < K) {                                   // clamp + select tail (K=3)
    const int k0 = kfull;
    V16 a;
#pragma unroll
    for (int j = 0; j < 4; ++j) {
      int kl0 = k0 + 2*j + 16*h,      kl1 = kl0 + 1;
      int kh0 = k0 + 8 + 2*j + 16*h, kh1 = kh0 + 1;
      float f0 = ap0[min(kl0, K-1) * ACS];
      float f1 = ap0[min(kl1, K-1) * ACS];
      float f2 = ap0[min(kh0, K-1) * ACS];
      float f3 = ap0[min(kh1, K-1) * ACS];
      f0 = (kl0 < K) ? f0 : 0.f;  f1 = (kl1 < K) ? f1 : 0.f;
      f2 = (kh0 < K) ? f2 : 0.f;  f3 = (kh1 < K) ? f3 : 0.f;
      a.p[j]     = __builtin_amdgcn_cvt_pkrtz(f0, f1);
      a.p[4 + j] = __builtin_amdgcn_cvt_pkrtz(f2, f3);
    }
    V16 bf[NT];
#pragma unroll
    for (int t = 0; t < NT; ++t)
#pragma unroll
      for (int j = 0; j < 8; ++j) {
        int kb0 = k0 + 2*j + 16*h, kb1 = kb0 + 1;
        float f0 = bp0[t][min(kb0, K-1) * BRS];
        float f1 = bp0[t][min(kb1, K-1) * BRS];
        f0 = (kb0 < K) ? f0 : 0.f;
        f1 = (kb1 < K) ? f1 : 0.f;
        bf[t].p[j] = __builtin_amdgcn_cvt_pkrtz(f0, f1);
      }
#pragma unroll
    for (int t = 0; t < NT; ++t)
      acc[t] = __builtin_amdgcn_wmma_f32_16x16x32_f16(false, a.v, false, bf[t].v,
                                                      (short)0, acc[t], false, false);
  }

  // store (bounds-checked; N multiple of 16 in all call sites, only M=50 odd)
#pragma unroll
  for (int t = 0; t < NT; ++t) {
    int n = n0 + 16 * t + r;
    if (n < N) {
#pragma unroll
      for (int rr = 0; rr < 8; ++rr) {
        int mm = m0 + rr + 8 * h;
        if (mm < M) {
          float o = acc[t][rr];
          if (bias) o += bias[mm];
          Db[mm * drs + n] = o;
        }
      }
    }
  }
}

// ---------------------------------------------------------------------------
// BatchNorm: stats over (batch, last) axes of a (Bn, C, Nn) tensor.
// ---------------------------------------------------------------------------
__global__ void k_bnstats(const float* __restrict__ pre, int Bn, int C, long long Nn,
                          const float* __restrict__ g, const float* __restrict__ beta,
                          float* __restrict__ scale, float* __restrict__ shift) {
  __shared__ float ssum[256], ssq[256];
  int c = blockIdx.x;
  long long cnt = (long long)Bn * Nn;
  float s = 0.f, q = 0.f;
  for (long long i = threadIdx.x; i < cnt; i += blockDim.x) {
    long long bp = i / Nn, nn = i % Nn;
    float v = pre[(bp * C + c) * Nn + nn];
    s += v; q += v * v;
  }
  ssum[threadIdx.x] = s; ssq[threadIdx.x] = q;
  __syncthreads();
  for (int st = 128; st > 0; st >>= 1) {
    if ((int)threadIdx.x < st) {
      ssum[threadIdx.x] += ssum[threadIdx.x + st];
      ssq[threadIdx.x]  += ssq[threadIdx.x + st];
    }
    __syncthreads();
  }
  if (threadIdx.x == 0) {
    float mean = ssum[0] / (float)cnt;
    float var  = ssq[0] / (float)cnt - mean * mean;
    float rstd = rsqrtf(var + 1e-5f);
    float sc = g[c] * rstd;
    scale[c] = sc;
    shift[c] = beta[c] - mean * sc;
  }
}

// act: 0 none, 1 relu, 2 leaky(0.2). Optional residual added AFTER activation.
__global__ void k_bnapply(const float* __restrict__ pre, const float* __restrict__ scale,
                          const float* __restrict__ shift, const float* __restrict__ res,
                          float* __restrict__ out, int C, long long Nn, long long total, int act) {
  long long i = (long long)blockIdx.x * blockDim.x + threadIdx.x;
  if (i >= total) return;
  int c = (int)((i / Nn) % C);
  float v = pre[i] * scale[c] + shift[c];
  if (act == 1)      v = v > 0.f ? v : 0.f;
  else if (act == 2) v = v > 0.f ? v : 0.2f * v;
  if (res) v += res[i];
  out[i] = v;
}

// ---------------------------------------------------------------------------
// Softmax over rows of energy[(b*N+n)][0..N-1]
// ---------------------------------------------------------------------------
__global__ void k_softmax(float* __restrict__ e, int Nn) {
  __shared__ float red[256];
  float* row = e + (long long)blockIdx.x * Nn;
  float mx = -3.4e38f;
  for (int i = threadIdx.x; i < Nn; i += 256) mx = fmaxf(mx, row[i]);
  red[threadIdx.x] = mx; __syncthreads();
  for (int st = 128; st > 0; st >>= 1) {
    if ((int)threadIdx.x < st) red[threadIdx.x] = fmaxf(red[threadIdx.x], red[threadIdx.x + st]);
    __syncthreads();
  }
  mx = red[0]; __syncthreads();
  float sm = 0.f;
  for (int i = threadIdx.x; i < Nn; i += 256) {
    float v = __expf(row[i] - mx); row[i] = v; sm += v;
  }
  red[threadIdx.x] = sm; __syncthreads();
  for (int st = 128; st > 0; st >>= 1) {
    if ((int)threadIdx.x < st) red[threadIdx.x] += red[threadIdx.x + st];
    __syncthreads();
  }
  float inv = 1.f / red[0];
  for (int i = threadIdx.x; i < Nn; i += 256) row[i] *= inv;
}

__global__ void k_colsum(const float* __restrict__ attn, float* __restrict__ cs, int Nn) {
  int t = blockIdx.x * blockDim.x + threadIdx.x;     // b*Nn + m
  if (t >= B8 * Nn) return;
  int b = t / Nn, m = t % Nn;
  const float* base = attn + (long long)b * Nn * Nn + m;
  float s = 0.f;
  for (int nn = 0; nn < Nn; ++nn) s += base[(long long)nn * Nn];
  cs[t] = s;
}

__global__ void k_colnorm(float* __restrict__ attn, const float* __restrict__ cs,
                          int Nn, long long total) {
  long long i = (long long)blockIdx.x * blockDim.x + threadIdx.x;
  if (i >= total) return;
  long long b = i / ((long long)Nn * Nn);
  int m = (int)(i % Nn);
  attn[i] /= (1e-9f + cs[b * Nn + m]);
}

__global__ void k_sub(const float* __restrict__ a, const float* __restrict__ b,
                      float* __restrict__ o, long long total) {
  long long i = (long long)blockIdx.x * blockDim.x + threadIdx.x;
  if (i < total) o[i] = a[i] - b[i];
}

// ---------------------------------------------------------------------------
// Furthest point sampling: one block per batch, xyz + dist resident in LDS.
// ---------------------------------------------------------------------------
__global__ void k_fps(const float* __restrict__ x, int* __restrict__ fidx) {
  __shared__ float xs[NPTS * 3];
  __shared__ float dist[NPTS];
  __shared__ float rv[256];
  __shared__ int   ri[256];
  __shared__ int   sfar;
  int b = blockIdx.x;
  const float* xb = x + (long long)b * 3 * NPTS;
  for (int n = threadIdx.x; n < NPTS; n += 256) {
    xs[n * 3 + 0] = xb[n];
    xs[n * 3 + 1] = xb[NPTS + n];
    xs[n * 3 + 2] = xb[2 * NPTS + n];
    dist[n] = 1e10f;
  }
  if (threadIdx.x == 0) sfar = 0;
  __syncthreads();
  for (int j = 0; j < NPOINT; ++j) {
    int far = sfar;
    if (threadIdx.x == 0) fidx[b * NPOINT + j] = far;
    float cx = xs[far * 3], cy = xs[far * 3 + 1], cz = xs[far * 3 + 2];
    float bmax = -1.f; int bidx = 0;
    for (int n = threadIdx.x; n < NPTS; n += 256) {
      float dx = xs[n * 3] - cx, dy = xs[n * 3 + 1] - cy, dz = xs[n * 3 + 2] - cz;
      float d = dx * dx + dy * dy + dz * dz;
      float dm = fminf(dist[n], d);
      dist[n] = dm;
      if (dm > bmax) { bmax = dm; bidx = n; }
    }
    rv[threadIdx.x] = bmax; ri[threadIdx.x] = bidx;
    __syncthreads();
    for (int st = 128; st > 0; st >>= 1) {
      if ((int)threadIdx.x < st) {
        float v2 = rv[threadIdx.x + st]; int i2 = ri[threadIdx.x + st];
        if (v2 > rv[threadIdx.x] || (v2 == rv[threadIdx.x] && i2 < ri[threadIdx.x])) {
          rv[threadIdx.x] = v2; ri[threadIdx.x] = i2;
        }
      }
      __syncthreads();
    }
    if (threadIdx.x == 0) sfar = ri[0];
    __syncthreads();
  }
}

// d2[b][s][n]
__global__ void k_d2(const float* __restrict__ x, const int* __restrict__ fidx,
                     float* __restrict__ d2, long long total) {
  long long i = (long long)blockIdx.x * 256 + threadIdx.x;
  if (i >= total) return;
  int n = (int)(i & (NPTS - 1));
  long long rest = i >> 11;
  int s = (int)(rest & (NPOINT - 1));
  int b = (int)(rest >> 9);
  const float* xb = x + (long long)b * 3 * NPTS;
  int c0 = fidx[b * NPOINT + s];
  float ax = xb[c0], ay = xb[NPTS + c0], az = xb[2 * NPTS + c0];
  float bx = xb[n],  by = xb[NPTS + n],  bz = xb[2 * NPTS + n];
  float aa = ax * ax + ay * ay + az * az;
  float bb = bx * bx + by * by + bz * bz;
  float ab = ax * bx + ay * by + az * bz;
  d2[i] = aa + bb - 2.f * ab;
}

// 32-NN selection per (b,s) row of d2, LDS-resident row.
__global__ void k_knn32(const float* __restrict__ d2, int* __restrict__ kidx) {
  __shared__ float vals[NPTS];
  __shared__ float rv[256];
  __shared__ int   ri[256];
  long long row = blockIdx.x;                       // b*NPOINT + s
  const float* src = d2 + row * NPTS;
  for (int n = threadIdx.x; n < NPTS; n += 256) vals[n] = src[n];
  __syncthreads();
  for (int j = 0; j < NSAMPLE; ++j) {
    float bv = 3.4e38f; int bi = 0;
    for (int n = threadIdx.x; n < NPTS; n += 256) {
      float v = vals[n];
      if (v < bv) { bv = v; bi = n; }
    }
    rv[threadIdx.x] = bv; ri[threadIdx.x] = bi;
    __syncthreads();
    for (int st = 128; st > 0; st >>= 1) {
      if ((int)threadIdx.x < st) {
        float v2 = rv[threadIdx.x + st]; int i2 = ri[threadIdx.x + st];
        if (v2 < rv[threadIdx.x] || (v2 == rv[threadIdx.x] && i2 < ri[threadIdx.x])) {
          rv[threadIdx.x] = v2; ri[threadIdx.x] = i2;
        }
      }
      __syncthreads();
    }
    if (threadIdx.x == 0) { kidx[row * NSAMPLE + j] = ri[0]; vals[ri[0]] = 3.4e38f; }
    __syncthreads();
  }
}

// 3-NN per (b,n) scanning d2[b][*][n] + normalized inverse-distance weights.
__global__ void k_knn3(const float* __restrict__ d2, int* __restrict__ nidx,
                       float* __restrict__ w3) {
  int t = blockIdx.x * 256 + threadIdx.x;           // b*NPTS + n
  if (t >= B8 * NPTS) return;
  int b = t >> 11, n = t & (NPTS - 1);
  const float* col = d2 + (long long)b * NPOINT * NPTS + n;
  float d0 = 3.4e38f, d1 = 3.4e38f, dd2 = 3.4e38f;
  int i0 = 0, i1 = 0, i2 = 0;
  for (int s = 0; s < NPOINT; ++s) {
    float v = col[(long long)s * NPTS];
    if (v < d0)        { dd2 = d1; i2 = i1; d1 = d0; i1 = i0; d0 = v; i0 = s; }
    else if (v < d1)   { dd2 = d1; i2 = i1; d1 = v;  i1 = s; }
    else if (v < dd2)  { dd2 = v;  i2 = s; }
  }
  float w0 = 1.f / (d0 + 1e-8f), w1 = 1.f / (d1 + 1e-8f), w2 = 1.f / (dd2 + 1e-8f);
  float inv = 1.f / (w0 + w1 + w2);
  long long o = (long long)t * 3;
  nidx[o] = i0; nidx[o + 1] = i1; nidx[o + 2] = i2;
  w3[o] = w0 * inv; w3[o + 1] = w1 * inv; w3[o + 2] = w2 * inv;
}

// Grouped feature tensor (4096 groups, 256 ch, 32 samples): [gnorm ; center]
__global__ void k_newfeat(const float* __restrict__ h, const int* __restrict__ fidx,
                          const int* __restrict__ kidx, float* __restrict__ nf,
                          long long total) {
  long long i = (long long)blockIdx.x * 256 + threadIdx.x;
  if (i >= total) return;
  int k = (int)(i & (NSAMPLE - 1));
  long long r = i >> 5;
  int c = (int)(r & 255);
  int g = (int)(r >> 8);
  int b = g >> 9, s = g & (NPOINT - 1);
  int ci = fidx[b * NPOINT + s];
  int cc = c & 127;
  const float* hb = h + ((long long)b * 128 + cc) * NPTS;
  float center = hb[ci];
  float val;
  if (c < 128) { int pi = kidx[(long long)g * NSAMPLE + k]; val = hb[pi] - center; }
  else         { val = center; }
  nf[i] = val;
}

__global__ void k_maxpool(const float* __restrict__ lo, float* __restrict__ feat, int total) {
  int t = blockIdx.x * 256 + threadIdx.x;           // g*256 + c
  if (t >= total) return;
  const float* p = lo + (long long)t * NSAMPLE;
  float m = p[0];
  for (int k = 1; k < NSAMPLE; ++k) m = fmaxf(m, p[k]);
  feat[t] = m;                                       // feat[b*512+s][c]
}

// cls path: conv(64x16) + bn over 8 batch samples + leaky(0.2).
__global__ void k_cls(const float* __restrict__ cls_label, const float* __restrict__ wl,
                      const float* __restrict__ gl, const float* __restrict__ bl,
                      float* __restrict__ cl) {
  int o = threadIdx.x;
  if (o >= 64) return;
  float raw[B8]; float mean = 0.f;
  for (int b = 0; b < B8; ++b) {
    float s = 0.f;
    for (int c = 0; c < 16; ++c) s += wl[o * 16 + c] * cls_label[b * 16 + c];
    raw[b] = s; mean += s;
  }
  mean *= (1.f / B8);
  float var = 0.f;
  for (int b = 0; b < B8; ++b) { float d = raw[b] - mean; var += d * d; }
  var *= (1.f / B8);
  float rstd = rsqrtf(var + 1e-5f);
  for (int b = 0; b < B8; ++b) {
    float v = (raw[b] - mean) * rstd * gl[o] + bl[o];
    v = v > 0.f ? v : 0.2f * v;
    cl[b * 64 + o] = v;
  }
}

__global__ void k_concat4(const float* __restrict__ x1, const float* __restrict__ x2,
                          const float* __restrict__ x3, const float* __restrict__ x4,
                          float* __restrict__ xcat, long long total) {
  long long i = (long long)blockIdx.x * 256 + threadIdx.x;
  if (i >= total) return;
  int n = (int)(i & (NPTS - 1));
  long long r = i >> 11;
  int c = (int)(r & 511);
  int b = (int)(r >> 9);
  const float* src = (c < 128) ? x1 : (c < 256) ? x2 : (c < 384) ? x3 : x4;
  int cc = c & 127;
  xcat[i] = src[((long long)b * 128 + cc) * NPTS + n];
}

// new_points rows 0..1087 = [cl(64) ; x_tr(1024)]
__global__ void k_points1(const float* __restrict__ cl, const float* __restrict__ xf,
                          float* __restrict__ np_, long long total) {
  long long i = (long long)blockIdx.x * 256 + threadIdx.x;
  if (i >= total) return;
  int n = (int)(i & (NPTS - 1));
  long long r = i >> 11;
  int c = (int)(r % 1088);
  int b = (int)(r / 1088);
  float v = (c < 64) ? cl[b * 64 + c]
                     : xf[((long long)b * 1024 + (c - 64)) * NPTS + n];
  np_[((long long)b * 1344 + c) * NPTS + n] = v;
}

// new_points rows 1088..1343 = 3-NN interpolated feature_0
__global__ void k_interp(const float* __restrict__ feat, const int* __restrict__ nidx,
                         const float* __restrict__ w3, float* __restrict__ np_,
                         long long total) {
  long long i = (long long)blockIdx.x * 256 + threadIdx.x;   // (b, c, n)
  if (i >= total) return;
  int n = (int)(i & (NPTS - 1));
  long long r = i >> 11;
  int c = (int)(r & 255);
  int b = (int)(r >> 8);
  long long t = ((long long)b * NPTS + n) * 3;
  float acc = 0.f;
  for (int j = 0; j < 3; ++j) {
    int s = nidx[t + j];
    acc += w3[t + j] * feat[((long long)b * NPOINT + s) * 256 + c];
  }
  np_[((long long)b * 1344 + 1088 + c) * NPTS + n] = acc;
}

// ---------------------------------------------------------------------------
// Host side
// ---------------------------------------------------------------------------
namespace {

struct SA { const float *wqk, *wv, *bv, *wt, *bt, *g, *b; };

// brs/acs must be one of the compiled combos: (2048,1), (2048,2048), (32,1)
inline void gemm(hipStream_t s,
                 const float* A, int ars, int acs, long long abs_,
                 const float* Bm, int brs, long long bbs,
                 float* D, int drs, long long dbs, const float* bias,
                 int M, int N, int K, int batches) {
  if (brs == NPTS && acs == 1) {
    dim3 g((N + 63) / 64, (M + 15) / 16, batches);
    k_gemm<4, NPTS, 1><<<g, 32, 0, s>>>(A, ars, abs_, Bm, bbs, D, drs, dbs, bias, M, N, K);
  } else if (brs == NPTS) {
    dim3 g((N + 63) / 64, (M + 15) / 16, batches);
    k_gemm<4, NPTS, NPTS><<<g, 32, 0, s>>>(A, ars, abs_, Bm, bbs, D, drs, dbs, bias, M, N, K);
  } else {
    dim3 g((N + 31) / 32, (M + 15) / 16, batches);
    k_gemm<2, NSAMPLE, 1><<<g, 32, 0, s>>>(A, ars, abs_, Bm, bbs, D, drs, dbs, bias, M, N, K);
  }
}

inline void bn(hipStream_t s, float* pre, int Bn, int C, long long Nn,
               const float* g, const float* beta, float* scale, float* shift,
               const float* res, float* out, int act) {
  k_bnstats<<<C, 256, 0, s>>>(pre, Bn, C, Nn, g, beta, scale, shift);
  long long total = (long long)Bn * C * Nn;
  k_bnapply<<<(unsigned)((total + 255) / 256), 256, 0, s>>>(pre, scale, shift,
                                                            res, out, C, Nn, total, act);
}

inline void sa_layer(hipStream_t s, const float* xin, float* xout, const SA& p,
                     float* q, float* v, float* xr, float* tmp,
                     float* energy, float* colsum, float* scale, float* shift) {
  const long long NN = (long long)NPTS * NPTS;
  // q = wqk x
  gemm(s, p.wqk, 128, 1, 0, xin, NPTS, 128LL * NPTS,
       q, NPTS, 32LL * NPTS, nullptr, 32, NPTS, 128, B8);
  // energy = q^T q  (A transposed: ars=1, acs=NPTS)
  gemm(s, q, 1, NPTS, 32LL * NPTS, q, NPTS, 32LL * NPTS,
       energy, NPTS, NN, nullptr, NPTS, NPTS, 32, B8);
  k_softmax<<<B8 * NPTS, 256, 0, s>>>(energy, NPTS);
  k_colsum<<<(B8 * NPTS + 255) / 256, 256, 0, s>>>(energy, colsum, NPTS);
  long long etot = (long long)B8 * NN;
  k_colnorm<<<(unsigned)((etot + 255) / 256), 256, 0, s>>>(energy, colsum, NPTS, etot);
  // v = wv x + bv
  gemm(s, p.wv, 128, 1, 0, xin, NPTS, 128LL * NPTS,
       v, NPTS, 128LL * NPTS, p.bv, 128, NPTS, 128, B8);
  // x_r = v attn
  gemm(s, v, NPTS, 1, 128LL * NPTS, energy, NPTS, NN,
       xr, NPTS, 128LL * NPTS, nullptr, 128, NPTS, NPTS, B8);
  long long tot = (long long)B8 * 128 * NPTS;
  k_sub<<<(unsigned)((tot + 255) / 256), 256, 0, s>>>(xin, xr, xr, tot);  // xr = x - x_r
  // t = wt (x - x_r) + bt
  gemm(s, p.wt, 128, 1, 0, xr, NPTS, 128LL * NPTS,
       tmp, NPTS, 128LL * NPTS, p.bt, 128, NPTS, 128, B8);
  // xout = x + relu(bn(t))
  bn(s, tmp, B8, 128, NPTS, p.g, p.b, scale, shift, xin, xout, 1);
}

} // namespace

extern "C" void kernel_launch(void* const* d_in, const int* in_sizes, int n_in,
                              void* d_out, int out_size, void* d_ws, size_t ws_size,
                              hipStream_t stream) {
  (void)in_sizes; (void)n_in; (void)out_size;
  auto P = [&](int i) { return (const float*)d_in[i]; };

  const float* x         = P(0);
  const float* cls_label = P(1);
  const float* w1 = P(2);  const float* g1 = P(3);  const float* b1 = P(4);
  const float* w2 = P(5);  const float* g2 = P(6);  const float* b2 = P(7);
  SA sa[4];
  for (int l = 0; l < 4; ++l) {
    int base = 8 + 7 * l;
    sa[l] = { P(base), P(base + 1), P(base + 2), P(base + 3),
              P(base + 4), P(base + 5), P(base + 6) };
  }
  const float* wf  = P(36); const float* gf  = P(37); const float* bf  = P(38);
  const float* wl  = P(39); const float* gl  = P(40); const float* bl  = P(41);
  const float* wloc= P(42); const float* gloc= P(43); const float* bloc= P(44);
  const float* wfp = P(45); const float* bfp = P(46);
  const float* gfp = P(47); const float* betafp = P(48);
  const float* ws1 = P(49); const float* bs1 = P(50);
  const float* gs1 = P(51); const float* betas1 = P(52);
  const float* ws2 = P(53); const float* bs2 = P(54);
  const float* gs2 = P(55); const float* betas2 = P(56);
  const float* ws3 = P(57); const float* bs3 = P(58);

  // ---- workspace arena ----
  char* base = (char*)d_ws;
  size_t cur = 0;
  auto alloc = [&](size_t nfloats) -> float* {
    float* p = (float*)(base + cur);
    cur += ((nfloats * sizeof(float)) + 255) & ~(size_t)255;
    return p;
  };
  const long long NN = (long long)NPTS * NPTS;
  float* h1     = alloc((size_t)B8 * 64 * NPTS);
  float* h      = alloc((size_t)B8 * 128 * NPTS);
  float* x1     = alloc((size_t)B8 * 128 * NPTS);
  float* x2     = alloc((size_t)B8 * 128 * NPTS);
  float* x3     = alloc((size_t)B8 * 128 * NPTS);
  float* x4     = alloc((size_t)B8 * 128 * NPTS);
  float* q      = alloc((size_t)B8 * 32 * NPTS);
  float* v      = alloc((size_t)B8 * 128 * NPTS);
  float* xr     = alloc((size_t)B8 * 128 * NPTS);
  float* tmp    = alloc((size_t)B8 * 128 * NPTS);
  float* energy = alloc((size_t)B8 * NN);        // reused: newfeat, later s2 pre
  float* colsum = alloc((size_t)B8 * NPTS);
  float* scale  = alloc(2048);
  float* shift  = alloc(2048);
  float* xcat   = alloc((size_t)B8 * 512 * NPTS);
  float* xf     = alloc((size_t)B8 * 1024 * NPTS);   // reused: fp
  int*   fidx   = (int*)alloc((size_t)B8 * NPOINT);
  float* d2buf  = alloc((size_t)B8 * NPOINT * NPTS);
  int*   kidx   = (int*)alloc((size_t)B8 * NPOINT * NSAMPLE);
  float* lo     = alloc((size_t)B8 * NPOINT * 256 * NSAMPLE);
  float* feat   = alloc((size_t)B8 * NPOINT * 256);
  float* cl     = alloc((size_t)B8 * 64);
  int*   nidx   = (int*)alloc((size_t)B8 * NPTS * 3);
  float* w3b    = alloc((size_t)B8 * NPTS * 3);
  float* npts_  = alloc((size_t)B8 * 1344 * NPTS);   // reused: s1 pre
  if (cur > ws_size) return;

  float* newfeat = energy;     // energy dead after SA layers
  float* fpbuf   = xf;         // xf dead once new_points is built
  float* s1buf   = npts_;      // new_points dead after fp GEMM
  float* s2buf   = energy;     // grouped tensor dead after local conv

  // ---- encoder ----
  gemm(stream, w1, 3, 1, 0, x, NPTS, 3LL * NPTS,
       h1, NPTS, 64LL * NPTS, nullptr, 64, NPTS, 3, B8);
  bn(stream, h1, B8, 64, NPTS, g1, b1, scale, shift, nullptr, h1, 1);
  gemm(stream, w2, 64, 1, 0, h1, NPTS, 64LL * NPTS,
       h, NPTS, 128LL * NPTS, nullptr, 128, NPTS, 64, B8);
  bn(stream, h, B8, 128, NPTS, g2, b2, scale, shift, nullptr, h, 1);

  sa_layer(stream, h,  x1, sa[0], q, v, xr, tmp, energy, colsum, scale, shift);
  sa_layer(stream, x1, x2, sa[1], q, v, xr, tmp, energy, colsum, scale, shift);
  sa_layer(stream, x2, x3, sa[2], q, v, xr, tmp, energy, colsum, scale, shift);
  sa_layer(stream, x3, x4, sa[3], q, v, xr, tmp, energy, colsum, scale, shift);

  long long ctot = (long long)B8 * 512 * NPTS;
  k_concat4<<<(unsigned)((ctot + 255) / 256), 256, 0, stream>>>(x1, x2, x3, x4, xcat, ctot);
  gemm(stream, wf, 512, 1, 0, xcat, NPTS, 512LL * NPTS,
       xf, NPTS, 1024LL * NPTS, nullptr, 1024, NPTS, 512, B8);
  bn(stream, xf, B8, 1024, NPTS, gf, bf, scale, shift, nullptr, xf, 2);

  // ---- sampling / grouping ----
  k_fps<<<B8, 256, 0, stream>>>(x, fidx);
  long long dtot = (long long)B8 * NPOINT * NPTS;
  k_d2<<<(unsigned)((dtot + 255) / 256), 256, 0, stream>>>(x, fidx, d2buf, dtot);
  k_knn32<<<B8 * NPOINT, 256, 0, stream>>>(d2buf, kidx);
  k_knn3<<<(B8 * NPTS + 255) / 256, 256, 0, stream>>>(d2buf, nidx, w3b);

  long long nftot = (long long)B8 * NPOINT * 256 * NSAMPLE;
  k_newfeat<<<(unsigned)((nftot + 255) / 256), 256, 0, stream>>>(h, fidx, kidx, newfeat, nftot);
  gemm(stream, wloc, 256, 1, 0, newfeat, NSAMPLE, 256LL * NSAMPLE,
       lo, NSAMPLE, 256LL * NSAMPLE, nullptr, 256, NSAMPLE, 256, B8 * NPOINT);
  bn(stream, lo, B8 * NPOINT, 256, NSAMPLE, gloc, bloc, scale, shift, nullptr, lo, 1);
  k_maxpool<<<(B8 * NPOINT * 256 + 255) / 256, 256, 0, stream>>>(lo, feat, B8 * NPOINT * 256);

  // ---- decoder ----
  k_cls<<<1, 64, 0, stream>>>(cls_label, wl, gl, bl, cl);
  long long p1tot = (long long)B8 * 1088 * NPTS;
  k_points1<<<(unsigned)((p1tot + 255) / 256), 256, 0, stream>>>(cl, xf, npts_, p1tot);
  long long itot = (long long)B8 * 256 * NPTS;
  k_interp<<<(unsigned)((itot + 255) / 256), 256, 0, stream>>>(feat, nidx, w3b, npts_, itot);

  gemm(stream, wfp, 1344, 1, 0, npts_, NPTS, 1344LL * NPTS,
       fpbuf, NPTS, 1024LL * NPTS, bfp, 1024, NPTS, 1344, B8);
  bn(stream, fpbuf, B8, 1024, NPTS, gfp, betafp, scale, shift, nullptr, fpbuf, 1);
  gemm(stream, ws1, 1024, 1, 0, fpbuf, NPTS, 1024LL * NPTS,
       s1buf, NPTS, 512LL * NPTS, bs1, 512, NPTS, 1024, B8);
  bn(stream, s1buf, B8, 512, NPTS, gs1, betas1, scale, shift, nullptr, s1buf, 1);
  gemm(stream, ws2, 512, 1, 0, s1buf, NPTS, 512LL * NPTS,
       s2buf, NPTS, 256LL * NPTS, bs2, 256, NPTS, 512, B8);
  bn(stream, s2buf, B8, 256, NPTS, gs2, betas2, scale, shift, nullptr, s2buf, 1);
  gemm(stream, ws3, 256, 1, 0, s2buf, NPTS, 256LL * NPTS,
       (float*)d_out, NPTS, (long long)PART * NPTS, bs3, PART, NPTS, 256, B8);
}
